// LimitRegressor_62526133895600
// MI455X (gfx1250) — matrix operand order repaired and checked by reading, not compiled, not run on hardware
//
#include <hip/hip_runtime.h>
#include <hip/hip_bf16.h>
#include <stdint.h>

// ---------------------------------------------------------------------------
// Types for CDNA5 WMMA (wave32): bf16 A/B fragments (16 x bf16 = 8 VGPRs),
// f32 C/D accumulator (8 VGPRs).
// ---------------------------------------------------------------------------
typedef __bf16 bf16_t;
typedef __attribute__((ext_vector_type(16))) bf16_t v16bf;
typedef __attribute__((ext_vector_type(8)))  float  v8f;
typedef __attribute__((ext_vector_type(4)))  uint32_t v4u;

// GCC-style vector matching the async-LDS builtin's parameter type.
typedef int v4i_g __attribute__((vector_size(16)));
typedef __attribute__((address_space(1))) v4i_g* gptr_b128;
typedef __attribute__((address_space(3))) v4i_g* lptr_b128;

union FragAB {
    v16bf bf;
    v4u   q[2];
};

static __device__ __forceinline__ uint16_t f32_to_bf16_rne(float f) {
    uint32_t b = __float_as_uint(f);
    uint32_t r = (b + 0x7FFFu + ((b >> 16) & 1u)) >> 16;
    return (uint16_t)r;
}

// ---------------------------------------------------------------------------
// CDNA5 async global->LDS copy (16 bytes per lane), ASYNCcnt-tracked.
// Falls back to a b128 load + b128 ds_store when the builtin is absent.
// ---------------------------------------------------------------------------
#if __has_builtin(__builtin_amdgcn_global_load_async_to_lds_b128)
#define HAVE_ASYNC_LDS 1
#else
#define HAVE_ASYNC_LDS 0
#endif

static __device__ __forceinline__ void copy16_async(uint16_t* lds,
                                                    const uint16_t* g) {
#if HAVE_ASYNC_LDS
    __builtin_amdgcn_global_load_async_to_lds_b128(
        (gptr_b128)(uint16_t*)g,   // global source (cast away const)
        (lptr_b128)lds,            // LDS destination
        0, 0);
#else
    *(v4u*)lds = *(const v4u*)g;
#endif
}

static __device__ __forceinline__ void async_wait_all() {
#if HAVE_ASYNC_LDS && __has_builtin(__builtin_amdgcn_s_wait_asynccnt)
    __builtin_amdgcn_s_wait_asynccnt(0);
#endif
}

// ---------------------------------------------------------------------------
// f32 -> bf16 conversion (grid-stride)
// ---------------------------------------------------------------------------
__global__ void cvt_f32_bf16_kernel(const float* __restrict__ in,
                                    uint16_t* __restrict__ out, int n) {
    int stride = gridDim.x * blockDim.x;
    for (int i = blockIdx.x * blockDim.x + threadIdx.x; i < n; i += stride)
        out[i] = f32_to_bf16_rne(in[i]);
}

// ---------------------------------------------------------------------------
// f32[K][N] -> bf16[N][K] transpose+convert (weights; done once, reused by
// every M-block, so GEMM B-tiles become contiguous 16B chunks).
// 32x32 tiles, 256 threads (32x8).
// ---------------------------------------------------------------------------
__global__ __launch_bounds__(256) void cvt_transpose_bf16_kernel(
    const float* __restrict__ in, uint16_t* __restrict__ out, int K, int N) {
    __shared__ float tile[32][33];
    const int kb = blockIdx.y * 32;
    const int nb = blockIdx.x * 32;
    const int tx = threadIdx.x & 31;
    const int ty = threadIdx.x >> 5;
#pragma unroll
    for (int i = ty; i < 32; i += 8)
        tile[i][tx] = in[(size_t)(kb + i) * N + nb + tx];
    __syncthreads();
#pragma unroll
    for (int i = ty; i < 32; i += 8)
        out[(size_t)(nb + i) * K + kb + tx] = f32_to_bf16_rne(tile[tx][i]);
}

// ---------------------------------------------------------------------------
// bf16 WMMA GEMM:  C[M,N] = act(A[M,K] @ Bt[N,K]^T + bias)
//   block = 256 threads = 8 waves; block tile 256(M) x 64(N); K chunk 32.
//   wave w computes rows [w*32, w*32+32) x 64 N columns: 2 M sub-tiles x
//   4 N sub-tiles = 8 WMMAs per K-step, each B fragment reused twice
//   (raises the WMMA : ds_load_b128 ratio to 8:12 from 4:10).
//   A and Bt tiles are contiguous in global -> staged to LDS with 16-byte
//   async copies (ASYNCcnt), double-buffered so copy(k+1) overlaps wmma(k).
//   LDS rows padded to 40 bf16 (80 B): 16B-aligned B128 lanes, spread banks.
//
// Fragment layouts per CDNA5 ISA 7.12.2 (16-bit, wave32), h = lane>>4:
//   A: lane -> row = lane&15, K = {8h..8h+7, 16+8h..+7}  (2 x b128 reads)
//   B: lane -> col = lane&15, K = {8h..8h+7, 16+8h..+7}  (2 x b128 reads)
//   C/D: vgpr d -> M = d + 8h, N = lane&15.
// ---------------------------------------------------------------------------
#define MT 256
#define NT 64
#define KT 32
#define MW 32     // M rows per wave (2 x 16 sub-tiles)
#define LDSS 40   // padded row stride in bf16 elements (80 bytes)

template <bool RELU, bool OUT_F32>
__global__ __launch_bounds__(256) void gemm_bf16_wmma(
    const uint16_t* __restrict__ A,   // [M][K] bf16 row-major
    const uint16_t* __restrict__ Bt,  // [N][K] bf16 row-major (B transposed)
    const float* __restrict__ bias, void* __restrict__ C,
    int M, int N, int K) {
    __shared__ __align__(16) uint16_t lA[2][MT * LDSS];  // 2 x 20480 B
    __shared__ __align__(16) uint16_t lB[2][NT * LDSS];  // 2 x  5120 B

    const int tid  = threadIdx.x;
    const int wave = tid >> 5;
    const int lane = tid & 31;
    const int half = lane >> 4;
    const int nl   = lane & 15;

    const int m0 = blockIdx.y * MT;
    const int n0 = blockIdx.x * NT;

    v8f acc[2][4] = {};

    auto stage = [&](int buf, int kc) {
        const int k0 = kc * KT;
        // A tile: 256 rows x 32 bf16 = 1024 x 16B chunks, 4 per thread
#pragma unroll
        for (int it = 0; it < 4; ++it) {
            const int c = tid + it * 256;
            const int r = c >> 2, s = (c & 3) * 8;
            copy16_async(&lA[buf][r * LDSS + s],
                         A + (size_t)(m0 + r) * K + k0 + s);
        }
        // B tile: 64 rows x 32 bf16 = 256 x 16B chunks, 1 per thread
        {
            const int r = tid >> 2, s = (tid & 3) * 8;
            copy16_async(&lB[buf][r * LDSS + s],
                         Bt + (size_t)(n0 + r) * K + k0 + s);
        }
    };

    const int nk = K / KT;
    stage(0, 0);

    for (int kc = 0; kc < nk; ++kc) {
        const int buf = kc & 1;
        async_wait_all();     // my async writes for `buf` have landed
        __syncthreads();      // everyone's have landed; prev reads long done
        if (kc + 1 < nk) stage(buf ^ 1, kc + 1);  // overlaps with wmma below

        FragAB fa[2];
#pragma unroll
        for (int t = 0; t < 2; ++t) {
            const uint16_t* base =
                &lA[buf][(wave * MW + t * 16 + nl) * LDSS + half * 8];
            fa[t].q[0] = *(const v4u*)(base);
            fa[t].q[1] = *(const v4u*)(base + 16);
        }
#pragma unroll
        for (int j = 0; j < 4; ++j) {
            FragAB fb;
            const uint16_t* bb = &lB[buf][(j * 16 + nl) * LDSS + half * 8];
            fb.q[0] = *(const v4u*)(bb);
            fb.q[1] = *(const v4u*)(bb + 16);
#pragma unroll
            for (int t = 0; t < 2; ++t)
                acc[t][j] = __builtin_amdgcn_wmma_f32_16x16x32_bf16(
                    false, fa[t].bf, false, fb.bf, (short)0, acc[t][j],
                    false, false);
        }
    }

    // --- epilogue: bias (+relu), store bf16 or f32 ---
#pragma unroll
    for (int j = 0; j < 4; ++j) {
        const int col = n0 + j * 16 + nl;
        const float bs = bias[col];
#pragma unroll
        for (int t = 0; t < 2; ++t) {
            const int mw = m0 + wave * MW + t * 16;
#pragma unroll
            for (int d = 0; d < 8; ++d) {
                const int row = mw + d + 8 * half;
                float v = acc[t][j][d] + bs;
                if (RELU) v = fmaxf(v, 0.f);
                if (OUT_F32)
                    ((float*)C)[(size_t)row * N + col] = v;
                else
                    ((uint16_t*)C)[(size_t)row * N + col] = f32_to_bf16_rne(v);
            }
        }
    }
}

// ---------------------------------------------------------------------------
// Fixed-point tail.  r <- r - 0.81*tanh(r) is monotone in r, so the element
// with max |out| always produces max |delta|; the global step count K* is
// obtained by simulating the scalar recurrence on max|out|.
// ---------------------------------------------------------------------------
__global__ void init_scalars_kernel(int* s) {
    s[0] = 0;  // max |out| as ordered float bits (values are >= 0)
    s[1] = 0;  // K*
}

__global__ void maxabs_kernel(const float* __restrict__ x, int n,
                              int* __restrict__ result) {
    __shared__ float red[256];
    float m = 0.f;
    int stride = gridDim.x * blockDim.x;
    for (int i = blockIdx.x * blockDim.x + threadIdx.x; i < n; i += stride)
        m = fmaxf(m, fabsf(x[i]));
    red[threadIdx.x] = m;
    __syncthreads();
    for (int s = 128; s > 0; s >>= 1) {
        if (threadIdx.x < s)
            red[threadIdx.x] = fmaxf(red[threadIdx.x], red[threadIdx.x + s]);
        __syncthreads();
    }
    if (threadIdx.x == 0)
        atomicMax(result, __float_as_int(red[0]));
}

__global__ void compute_k_kernel(const int* __restrict__ maxbits,
                                 int* __restrict__ kout) {
    float r = __int_as_float(maxbits[0]);
    int K = 0;
    for (int k = 1; k <= 100; ++k) {
        float d = 0.81f * tanhf(r);  // ALPHA * PHI_C = 0.81
        r -= d;
        K = k;
        if (d < 1e-6f) break;  // converging step IS applied
    }
    kout[0] = K;
}

__global__ void fixpoint_kernel(float* __restrict__ out, int n,
                                const int* __restrict__ kptr) {
    const int K = kptr[0];
    int stride = gridDim.x * blockDim.x;
    for (int i = blockIdx.x * blockDim.x + threadIdx.x; i < n; i += stride) {
        const float o = out[i];
        float y = 0.f;
        for (int k = 0; k < K; ++k)
            y += 0.81f * tanhf(o - y);
        out[i] = y;
    }
}

// ---------------------------------------------------------------------------
// Launcher.  inputs: x[16384,1024] W1[1024,4096] b1[4096] W2[4096,1024] b2[1024]
// ws layout: [0..255] scalars | xb bf16 | W1t bf16 | W2t bf16 | h bf16
// ---------------------------------------------------------------------------
extern "C" void kernel_launch(void* const* d_in, const int* in_sizes, int n_in,
                              void* d_out, int out_size, void* d_ws, size_t ws_size,
                              hipStream_t stream) {
    (void)n_in; (void)ws_size;
    const float* x  = (const float*)d_in[0];
    const float* W1 = (const float*)d_in[1];
    const float* b1 = (const float*)d_in[2];
    const float* W2 = (const float*)d_in[3];
    const float* b2 = (const float*)d_in[4];

    const int DIN = 1024, DH = 4096, DOUT = 1024;
    const int Bm = in_sizes[0] / DIN;  // 16384

    char* ws = (char*)d_ws;
    int* scal = (int*)ws;
    uint16_t* xb  = (uint16_t*)(ws + 256);
    uint16_t* w1t = xb  + (size_t)Bm * DIN;   // [DH][DIN]
    uint16_t* w2t = w1t + (size_t)DIN * DH;   // [DOUT][DH]
    uint16_t* h   = w2t + (size_t)DH * DOUT;  // [Bm][DH]
    float* out = (float*)d_out;

    // 1) conversions (+ one-time weight transposes for contiguous B tiles)
    cvt_f32_bf16_kernel<<<1024, 256, 0, stream>>>(x, xb, Bm * DIN);
    {
        dim3 g1(DH / 32, DIN / 32);
        cvt_transpose_bf16_kernel<<<g1, 256, 0, stream>>>(W1, w1t, DIN, DH);
        dim3 g2(DOUT / 32, DH / 32);
        cvt_transpose_bf16_kernel<<<g2, 256, 0, stream>>>(W2, w2t, DH, DOUT);
    }

    // 2) h = relu(x @ W1 + b1)  -> bf16
    {
        dim3 grid(DH / NT, Bm / MT);
        gemm_bf16_wmma<true, false><<<grid, 256, 0, stream>>>(
            xb, w1t, b1, (void*)h, Bm, DH, DIN);
    }
    // 3) out = h @ W2 + b2  -> f32 (into d_out)
    {
        dim3 grid(DOUT / NT, Bm / MT);
        gemm_bf16_wmma<false, true><<<grid, 256, 0, stream>>>(
            h, w2t, b2, (void*)out, Bm, DOUT, DH);
    }

    // 4) fixed point: global K* from max|out|, then K* steps per element
    init_scalars_kernel<<<1, 1, 0, stream>>>(scal);
    maxabs_kernel<<<1024, 256, 0, stream>>>(out, out_size, scal);
    compute_k_kernel<<<1, 1, 0, stream>>>(scal, scal + 1);
    fixpoint_kernel<<<2048, 256, 0, stream>>>(out, out_size, scal + 1);
}
